// GNN_Final_VN_Model_58385785422523
// MI455X (gfx1250) — compile-verified, matrix operand order checked by reading, not run on hardware
//
#include <hip/hip_runtime.h>

#define THREADS 256

typedef __attribute__((ext_vector_type(16))) __bf16 v16bf;
typedef __attribute__((ext_vector_type(8)))  float  v8f;

union FragAB { v16bf v; unsigned u[8]; };

__device__ __forceinline__ unsigned short f2bf(float f) {
    unsigned u = __float_as_uint(f);
    unsigned r = u + 0x7fffu + ((u >> 16) & 1u);   // round-to-nearest-even
    return (unsigned short)(r >> 16);
}

// order-preserving float->uint key for atomicMax on floats
__device__ __forceinline__ unsigned fenc(float f) {
    unsigned u = __float_as_uint(f);
    return (u & 0x80000000u) ? ~u : (u | 0x80000000u);
}
__device__ __forceinline__ float fdec(unsigned k) {
    unsigned u = (k & 0x80000000u) ? (k ^ 0x80000000u) : ~k;
    return __uint_as_float(u);
}

// ---------------- conversion kernels ----------------

__global__ void k_cvt_bf16(const float* __restrict__ x, unsigned short* __restrict__ xb, int n) {
    int t = blockIdx.x * blockDim.x + threadIdx.x;
    if (t < n) xb[t] = f2bf(x[t]);
}

// W [K][C] f32  ->  Wt [C][K] bf16  (transpose so B-fragment K is contiguous)
__global__ void k_wt(const float* __restrict__ W, unsigned short* __restrict__ Wt, int K, int C) {
    int t = blockIdx.x * blockDim.x + threadIdx.x;
    if (t >= K * C) return;
    int c = t % C, k = t / C;
    Wt[(size_t)c * K + k] = f2bf(W[(size_t)k * C + c]);
}

__global__ void k_zero_f32(float* p, int n) {
    int t = blockIdx.x * blockDim.x + threadIdx.x;
    if (t < n) p[t] = 0.0f;
}

__global__ void k_init_attn(unsigned* mkey, float* den, int n) {
    int t = blockIdx.x * blockDim.x + threadIdx.x;
    if (t < n) { mkey[t] = 0u; den[t] = 0.0f; }
}

// ---------------- WMMA bf16 GEMM:  C[M][Cc] = A[M][K] * Bt[Cc][K]^T (+bias) --------------
// One wave computes a 16x64 strip (4 col tiles): A fragment loaded once per k-step and
// reused across 4 v_wmma_f32_16x16x32_bf16 issues -> 4x less A traffic, 4x wmma density.

__global__ void k_gemm_bf16(const unsigned short* __restrict__ A,
                            const unsigned short* __restrict__ Bt,
                            float* __restrict__ Cm,
                            const float* __restrict__ bias,
                            int M, int K, int Cc) {
    int wave = (blockIdx.x * blockDim.x + threadIdx.x) >> 5;
    int lane = threadIdx.x & 31;
    int ngroups = Cc >> 6;               // groups of 4 col-tiles (64 cols)
    int mt = wave / ngroups;
    int g  = wave % ngroups;
    if (mt * 16 >= M) return;

    int half = lane >> 4;     // K-half selector (ISA 16-bit A/B layout)
    int lid  = lane & 15;     // row (A) / col (B) index within tile

    const unsigned short* arow = A + (size_t)(mt * 16 + lid) * K;
    const unsigned short* bc0 = Bt + (size_t)(g * 64 +  0 + lid) * K;
    const unsigned short* bc1 = Bt + (size_t)(g * 64 + 16 + lid) * K;
    const unsigned short* bc2 = Bt + (size_t)(g * 64 + 32 + lid) * K;
    const unsigned short* bc3 = Bt + (size_t)(g * 64 + 48 + lid) * K;

    v8f acc0 = {}, acc1 = {}, acc2 = {}, acc3 = {};
    for (int kb = 0; kb < K; kb += 32) {
        FragAB a, b0, b1, b2, b3;
#pragma unroll
        for (int p = 0; p < 8; ++p) {
            // pair p -> K offset per ISA 16-bit fragment layout:
            // vgpr0..3 hold K 0..7 (+half*8), vgpr4..7 hold K 16..23 (+half*8)
            int koff = kb + half * 8 + ((p < 4) ? (2 * p) : (16 + 2 * (p - 4)));
            a.u[p]  = *(const unsigned*)(arow + koff);
            b0.u[p] = *(const unsigned*)(bc0 + koff);
            b1.u[p] = *(const unsigned*)(bc1 + koff);
            b2.u[p] = *(const unsigned*)(bc2 + koff);
            b3.u[p] = *(const unsigned*)(bc3 + koff);
        }
        acc0 = __builtin_amdgcn_wmma_f32_16x16x32_bf16(false, a.v, false, b0.v, (short)0, acc0, false, false);
        acc1 = __builtin_amdgcn_wmma_f32_16x16x32_bf16(false, a.v, false, b1.v, (short)0, acc1, false, false);
        acc2 = __builtin_amdgcn_wmma_f32_16x16x32_bf16(false, a.v, false, b2.v, (short)0, acc2, false, false);
        acc3 = __builtin_amdgcn_wmma_f32_16x16x32_bf16(false, a.v, false, b3.v, (short)0, acc3, false, false);
    }

#pragma unroll
    for (int t = 0; t < 4; ++t) {
        v8f acc = (t == 0) ? acc0 : (t == 1) ? acc1 : (t == 2) ? acc2 : acc3;
        int col = g * 64 + t * 16 + lid;
        float badd = bias ? bias[col] : 0.0f;
#pragma unroll
        for (int r = 0; r < 8; ++r) {
            int row = mt * 16 + half * 8 + r;  // C/D layout: vgpr r -> M=r (+8 upper lanes)
            Cm[(size_t)row * Cc + col] = acc[r] + badd;
        }
    }
}

// ---------------- per-node attention scalars: es = h.a_s, ed = h.a_d (wave per node) -----

__global__ void k_rowdot2(const float* __restrict__ h, const float* __restrict__ as_,
                          const float* __restrict__ ad_, float* __restrict__ es,
                          float* __restrict__ ed, int N) {
    int wave = (blockIdx.x * blockDim.x + threadIdx.x) >> 5;
    int lane = threadIdx.x & 31;
    if (wave >= N) return;
    const float* row = h + (size_t)wave * 256;
    float s = 0.f, d = 0.f;
#pragma unroll
    for (int j = 0; j < 8; ++j) {
        int c = j * 32 + lane;          // coalesced across lanes
        float v = row[c];
        s += v * as_[c];
        d += v * ad_[c];
    }
#pragma unroll
    for (int m = 16; m >= 1; m >>= 1) {
        s += __shfl_xor(s, m, 32);
        d += __shfl_xor(d, m, 32);
    }
    if (lane == 0) { es[wave] = s; ed[wave] = d; }
}

// ---------------- edge kernels (ET = E + N self-loops) ----------------

__global__ void k_edge_max(const int* __restrict__ src, const int* __restrict__ dst,
                           const float* __restrict__ es, const float* __restrict__ ed,
                           float* __restrict__ ev, unsigned* __restrict__ mkey,
                           int E, int ET) {
    int e = blockIdx.x * blockDim.x + threadIdx.x;
    if (e >= ET) return;
    int s = (e < E) ? src[e] : (e - E);
    int d = (e < E) ? dst[e] : (e - E);
    float x = es[s] + ed[d];
    x = (x > 0.f) ? x : 0.2f * x;       // LeakyReLU(0.2)
    ev[e] = x;
    atomicMax(&mkey[d], fenc(x));
}

__global__ void k_edge_exp(const int* __restrict__ src, const int* __restrict__ dst,
                           float* __restrict__ ev, const unsigned* __restrict__ mkey,
                           float* __restrict__ den, int E, int ET) {
    int e = blockIdx.x * blockDim.x + threadIdx.x;
    if (e >= ET) return;
    int d = (e < E) ? dst[e] : (e - E);
    float w = __expf(ev[e] - fdec(mkey[d]));
    ev[e] = w;
    atomicAdd(&den[d], w);
}

// wave per edge; lanes cover channels c = j*32+lane (coalesced atomics, L2-resident)
__global__ void k_edge_scatter(const int* __restrict__ src, const int* __restrict__ dst,
                               const float* __restrict__ ev, const float* __restrict__ den,
                               const float* __restrict__ h, float* __restrict__ outb,
                               int E, int ET) {
    int wave = (blockIdx.x * blockDim.x + threadIdx.x) >> 5;
    int lane = threadIdx.x & 31;
    if (wave >= ET) return;
    int s = (wave < E) ? src[wave] : (wave - E);
    int d = (wave < E) ? dst[wave] : (wave - E);
    float alpha = ev[wave] / den[d];
    const float* hr = h + (size_t)s * 256;
    float* orow = outb + (size_t)d * 256;
#pragma unroll
    for (int j = 0; j < 8; ++j) {
        int c = j * 32 + lane;
        atomicAdd(&orow[c], hr[c] * alpha);
    }
}

// bias (+ optional LeakyReLU 0.01) and write bf16 copy for next layer's GEMM input
__global__ void k_finalize(float* __restrict__ outb, const float* __restrict__ bias,
                           unsigned short* __restrict__ xb, int total, int lrelu) {
    int t = blockIdx.x * blockDim.x + threadIdx.x;
    if (t >= total) return;
    int c = t & 255;
    float v = outb[t] + bias[c];
    if (lrelu) v = (v > 0.f) ? v : 0.01f * v;
    outb[t] = v;
    xb[t] = f2bf(v);
}

// ---------------- virtual node ----------------

__global__ void k_vn_reduce(const float* __restrict__ outb, float* __restrict__ vns, int N) {
    int c = threadIdx.x;                 // 256 channels
    int r0 = blockIdx.x * 256;
    int r1 = r0 + 256; if (r1 > N) r1 = N;
    float acc = 0.f;
    for (int r = r0; r < r1; ++r) acc += outb[(size_t)r * 256 + c];
    atomicAdd(&vns[c], acc);
}

__global__ void k_vn_mlp(const float* __restrict__ vns, const float* __restrict__ emb,
                         const float* W10, const float* b10, const float* W20, const float* b20,
                         const float* W11, const float* b11, const float* W21, const float* b21,
                         const float* W12, const float* b12, const float* W22, const float* b22,
                         float* __restrict__ out) {
    __shared__ float v[256];
    __shared__ float t[256];
    int tid = threadIdx.x;
    v[tid] = vns[tid] + emb[tid];
    __syncthreads();
    const float* W1s[3] = { W10, W11, W12 };
    const float* b1s[3] = { b10, b11, b12 };
    const float* W2s[3] = { W20, W21, W22 };
    const float* b2s[3] = { b20, b21, b22 };
    const int douts[3] = { 256, 256, 128 };
    for (int i = 0; i < 3; ++i) {
        float acc = b1s[i][tid];
        for (int k = 0; k < 256; ++k) acc += v[k] * W1s[i][k * 256 + tid];
        t[tid] = fmaxf(acc, 0.f);
        __syncthreads();
        int dco = douts[i];
        float a2 = 0.f;
        if (tid < dco) {
            a2 = b2s[i][tid];
            for (int k = 0; k < 256; ++k) a2 += t[k] * W2s[i][k * dco + tid];
        }
        __syncthreads();
        if (tid < dco) v[tid] = fmaxf(a2, 0.f);
        __syncthreads();
    }
    if (tid < 128) out[tid] = v[tid];
}

// ---------------- host side ----------------

static inline int cdiv(int a, int b) { return (a + b - 1) / b; }

extern "C" void kernel_launch(void* const* d_in, const int* in_sizes, int n_in,
                              void* d_out, int out_size, void* d_ws, size_t ws_size,
                              hipStream_t stream) {
    const float* x      = (const float*)d_in[0];
    const int*   ei     = (const int*)d_in[1];
    const float* vn_emb = (const float*)d_in[2];
    const float* W[4]  = { (const float*)d_in[3],  (const float*)d_in[7],
                           (const float*)d_in[11], (const float*)d_in[15] };
    const float* as_[4] = { (const float*)d_in[4],  (const float*)d_in[8],
                            (const float*)d_in[12], (const float*)d_in[16] };
    const float* ad_[4] = { (const float*)d_in[5],  (const float*)d_in[9],
                            (const float*)d_in[13], (const float*)d_in[17] };
    const float* b_[4]  = { (const float*)d_in[6],  (const float*)d_in[10],
                            (const float*)d_in[14], (const float*)d_in[18] };
    const float* m0W1 = (const float*)d_in[19]; const float* m0b1 = (const float*)d_in[20];
    const float* m0W2 = (const float*)d_in[21]; const float* m0b2 = (const float*)d_in[22];
    const float* m1W1 = (const float*)d_in[23]; const float* m1b1 = (const float*)d_in[24];
    const float* m1W2 = (const float*)d_in[25]; const float* m1b2 = (const float*)d_in[26];
    const float* m2W1 = (const float*)d_in[27]; const float* m2b1 = (const float*)d_in[28];
    const float* m2W2 = (const float*)d_in[29]; const float* m2b2 = (const float*)d_in[30];
    const float* Wout = (const float*)d_in[31]; const float* bout = (const float*)d_in[32];

    const int N  = in_sizes[0] / 128;
    const int E  = in_sizes[1] / 2;
    const int ET = E + N;                 // + self loops
    const int* srcp = ei;
    const int* dstp = ei + E;

    // workspace carve-up (256B aligned)
    char* ws = (char*)d_ws;
    size_t off = 0;
    auto take = [&](size_t bytes) -> char* {
        char* p = ws + off;
        off = (off + bytes + 255) & ~(size_t)255;
        return p;
    };
    unsigned short* xb   = (unsigned short*)take((size_t)N * 256 * 2);
    unsigned short* wt   = (unsigned short*)take((size_t)256 * 256 * 2);
    float*          h    = (float*)take((size_t)N * 256 * 4);
    float*          outb = (float*)take((size_t)N * 256 * 4);
    float*          es   = (float*)take((size_t)N * 4);
    float*          ed   = (float*)take((size_t)N * 4);
    float*          ev   = (float*)take((size_t)ET * 4);
    unsigned*       mkey = (unsigned*)take((size_t)N * 4);
    float*          den  = (float*)take((size_t)N * 4);
    float*          vns  = (float*)take((size_t)256 * 4);
    (void)ws_size; (void)n_in; (void)out_size;

    // x -> bf16 (layer-0 GEMM input, [N][128])
    k_cvt_bf16<<<cdiv(N * 128, THREADS), THREADS, 0, stream>>>(x, xb, N * 128);

    for (int l = 0; l < 4; ++l) {
        const int K = (l == 0) ? 128 : 256;

        // weight -> bf16 transposed [256][K]
        k_wt<<<cdiv(K * 256, THREADS), THREADS, 0, stream>>>(W[l], wt, K, 256);

        // h = x @ W  (WMMA bf16 -> f32), 16x64 strip per wave
        {
            int waves = cdiv(N, 16) * (256 / 64);
            k_gemm_bf16<<<cdiv(waves, THREADS / 32), THREADS, 0, stream>>>(
                xb, wt, h, nullptr, N, K, 256);
        }

        // es/ed per node
        k_rowdot2<<<cdiv(N * 32, THREADS), THREADS, 0, stream>>>(h, as_[l], ad_[l], es, ed, N);

        // init attention state + output accumulator
        k_init_attn<<<cdiv(N, THREADS), THREADS, 0, stream>>>(mkey, den, N);
        k_zero_f32<<<cdiv(N * 256, THREADS), THREADS, 0, stream>>>(outb, N * 256);

        // softmax over incoming edges + weighted aggregation
        k_edge_max<<<cdiv(ET, THREADS), THREADS, 0, stream>>>(srcp, dstp, es, ed, ev, mkey, E, ET);
        k_edge_exp<<<cdiv(ET, THREADS), THREADS, 0, stream>>>(srcp, dstp, ev, mkey, den, E, ET);
        k_edge_scatter<<<cdiv(ET * 32, THREADS), THREADS, 0, stream>>>(
            srcp, dstp, ev, den, h, outb, E, ET);

        // + bias, LeakyReLU(0.01) for layers 1..3, emit bf16 next-layer input
        k_finalize<<<cdiv(N * 256, THREADS), THREADS, 0, stream>>>(
            outb, b_[l], xb, N * 256, (l > 0) ? 1 : 0);
    }

    // virtual node: global add pool
    k_zero_f32<<<1, 256, 0, stream>>>(vns, 256);
    k_vn_reduce<<<cdiv(N, 256), 256, 0, stream>>>(outb, vns, N);

    // node_out = out @ Wout + bout   (WMMA, cols = 128)
    k_wt<<<cdiv(256 * 128, THREADS), THREADS, 0, stream>>>(Wout, wt, 256, 128);
    {
        int waves = cdiv(N, 16) * (128 / 64);
        k_gemm_bf16<<<cdiv(waves, THREADS / 32), THREADS, 0, stream>>>(
            xb, wt, (float*)d_out, bout, N, 256, 128);
    }

    // virtual-node MLP chain -> d_out tail
    k_vn_mlp<<<1, 256, 0, stream>>>(vns, vn_emb,
                                    m0W1, m0b1, m0W2, m0b2,
                                    m1W1, m1b1, m1W2, m1b2,
                                    m2W1, m2b1, m2W2, m2b2,
                                    (float*)d_out + (size_t)N * 128);
}
